// Attention_70660801953862
// MI455X (gfx1250) — matrix operand order repaired and checked by reading, not compiled, not run on hardware
//
#include <hip/hip_runtime.h>
#include <hip/hip_bf16.h>

// ---------------------------------------------------------------------------
// MI455X (gfx1250) fused attention:  x@Wqkv+b -> 8-head attention -> @Wout+b
// v_wmma_f32_16x16x32_bf16 everywhere; async global->LDS double buffering;
// DPP row-rotate butterfly reductions for the online softmax.
// ---------------------------------------------------------------------------

typedef __attribute__((ext_vector_type(16))) __bf16 v16bf;
typedef __attribute__((ext_vector_type(8)))  __bf16 v8bf;
typedef __attribute__((ext_vector_type(8)))  float  v8f;

#define DIM     512
#define NSEQ    1024
#define NHEAD   8
#define DHEAD   64
#define NBATCH  8                 // bt * b_sz
#define MROWS   (NBATCH * NSEQ)   // 8192
#define N_QKV   3072

// load 16 bf16 as an A/B fragment from two 16-byte chunks (16B aligned)
__device__ __forceinline__ v16bf load16(const __bf16* p0, const __bf16* p1) {
    v8bf a = *(const v8bf*)p0;
    v8bf b = *(const v8bf*)p1;
    v16bf r;
#pragma unroll
    for (int i = 0; i < 8; ++i) { r[i] = a[i]; r[i + 8] = b[i]; }
    return r;
}

__device__ __forceinline__ v8f wmma_bf16(v16bf a, v16bf b, v8f c) {
    return __builtin_amdgcn_wmma_f32_16x16x32_bf16(false, a, false, b,
                                                   (short)0, c, false, false);
}

// async copy 16 bytes global -> LDS (ASYNCcnt); lds offset = low 32b of ptr
__device__ __forceinline__ void async_ld16(void* lds, const void* g) {
    unsigned off = (unsigned)(unsigned long long)lds;
    asm volatile("global_load_async_to_lds_b128 %0, %1, off"
                 :: "v"(off), "v"(g) : "memory");
}
__device__ __forceinline__ void wait_async0() {
    asm volatile("s_wait_asynccnt 0x0" ::: "memory");
}

// DPP16 row-rotate (within 16-lane row) butterfly reductions — pure VALU.
template <int CTRL>
__device__ __forceinline__ float dpp_rot(float v) {
    return __int_as_float(__builtin_amdgcn_update_dpp(
        0, __float_as_int(v), CTRL, 0xf, 0xf, true));
}
__device__ __forceinline__ float row_max16(float v) {
    v = fmaxf(v, dpp_rot<0x121>(v));
    v = fmaxf(v, dpp_rot<0x122>(v));
    v = fmaxf(v, dpp_rot<0x124>(v));
    v = fmaxf(v, dpp_rot<0x128>(v));
    return v;
}
__device__ __forceinline__ float row_sum16(float v) {
    v += dpp_rot<0x121>(v);
    v += dpp_rot<0x122>(v);
    v += dpp_rot<0x124>(v);
    v += dpp_rot<0x128>(v);
    return v;
}

// ---------------------------------------------------------------------------
__global__ void __launch_bounds__(256) cvt_bf16_kernel(const float* __restrict__ in,
                                                       __bf16* __restrict__ out, int n) {
    int i = blockIdx.x * blockDim.x + threadIdx.x;
    if (i < n) out[i] = (__bf16)in[i];
}

// out[n*K + k] = (bf16) in[k*N + n]
__global__ void __launch_bounds__(256) transpose_bf16_kernel(const float* __restrict__ in,
                                                             __bf16* __restrict__ out,
                                                             int K, int N) {
    int i = blockIdx.x * blockDim.x + threadIdx.x;
    if (i < K * N) {
        int n = i / K;
        int k = i % K;
        out[i] = (__bf16)in[k * N + n];
    }
}

// ---------------------------------------------------------------------------
// GEMM: block tile 128x128, 8 waves, wave tile 32x64, K steps of 32,
// A/B tiles async-staged into LDS (double buffered), k-loop fully unrolled
// so buffer indices / LDS addresses are compile-time immediates.
// KIND 0: QKV epilogue (bias + scatter to Q/K/Vt)   KIND 1: f32 out + bias
template <int KIND>
__global__ void __launch_bounds__(256) gemm_kernel(
        const __bf16* __restrict__ A,       // [M][512]
        const __bf16* __restrict__ wT,      // [N][512]
        const float*  __restrict__ bias,    // [N]
        __bf16* __restrict__ Q, __bf16* __restrict__ Kt, __bf16* __restrict__ Vt,
        float* __restrict__ out)
{
    __shared__ __align__(16) __bf16 As[2][128][32];   // 16 KB
    __shared__ __align__(16) __bf16 Bs[2][128][32];   // 16 KB

    const int lane = threadIdx.x & 31;
    const int wave = threadIdx.x >> 5;
    const int rg = wave & 3;
    const int cg = wave >> 2;
    const int rowBase = blockIdx.x * 128;
    const int colBase = blockIdx.y * 128;
    const int lh = (lane < 16) ? 0 : 1;

#pragma unroll
    for (int i = 0; i < 2; ++i) {
        const int c = threadIdx.x + i * 256;        // chunk = 16B
        async_ld16((char*)(&As[0][0][0]) + c * 16,
                   A + (rowBase + (c >> 2)) * DIM + (c & 3) * 8);
        async_ld16((char*)(&Bs[0][0][0]) + c * 16,
                   wT + (colBase + (c >> 2)) * DIM + (c & 3) * 8);
    }

    v8f acc[2][4] = {};
#pragma unroll
    for (int k0 = 0; k0 < DIM; k0 += 32) {
        const int buf = (k0 >> 5) & 1;
        wait_async0();
        __syncthreads();
        if (k0 + 32 < DIM) {
            const int nb = buf ^ 1;
            const int k1 = k0 + 32;
#pragma unroll
            for (int i = 0; i < 2; ++i) {
                const int c = threadIdx.x + i * 256;
                async_ld16((char*)(&As[nb][0][0]) + c * 16,
                           A + (rowBase + (c >> 2)) * DIM + k1 + (c & 3) * 8);
                async_ld16((char*)(&Bs[nb][0][0]) + c * 16,
                           wT + (colBase + (c >> 2)) * DIM + k1 + (c & 3) * 8);
            }
        }
        // hoist ALL fragment loads so they clause; then 8 WMMAs back-to-back
        const __bf16* ab = &As[buf][rg * 32 + (lane & 15)][0];
        const v16bf a0 = load16(ab + lh * 8,           ab + 16 + lh * 8);
        const v16bf a1 = load16(ab + 16 * 32 + lh * 8, ab + 16 * 32 + 16 + lh * 8);
        v16bf bfr[4];
#pragma unroll
        for (int t = 0; t < 4; ++t) {
            const __bf16* bb = &Bs[buf][cg * 64 + t * 16 + (lane & 15)][0] + lh * 8;
            bfr[t] = load16(bb, bb + 16);
        }
#pragma unroll
        for (int t = 0; t < 4; ++t) {
            acc[0][t] = wmma_bf16(a0, bfr[t], acc[0][t]);
            acc[1][t] = wmma_bf16(a1, bfr[t], acc[1][t]);
        }
    }

    // ---- epilogue ----
#pragma unroll
    for (int t = 0; t < 4; ++t) {
        const int col = colBase + cg * 64 + t * 16 + (lane & 15);
        const float bv = bias[col];
#pragma unroll
        for (int rs = 0; rs < 2; ++rs) {
#pragma unroll
            for (int r = 0; r < 8; ++r) {
                const int m = rowBase + rg * 32 + rs * 16 + r + lh * 8;
                const float v = acc[rs][t][r] + bv;
                if constexpr (KIND == 0) {
                    const int sect = col >> 9;          // 0=q 1=k 2=v
                    const int h = (col >> 6) & 7;
                    const int d = col & 63;
                    const int bh = (m >> 10) * NHEAD + h;
                    const int nr = m & 1023;
                    if (sect == 0)      Q [((bh << 10) + nr) * DHEAD + d] = (__bf16)(v * 0.125f);
                    else if (sect == 1) Kt[((bh << 10) + nr) * DHEAD + d] = (__bf16)v;
                    else                Vt[((bh * DHEAD + d) << 10) + nr] = (__bf16)v;
                } else {
                    out[m * DIM + col] = v;
                }
            }
        }
    }
}

// ---------------------------------------------------------------------------
// Flash attention: block = 256 threads (8 waves x 16 query rows = 128 rows),
// K/V blocks (32 keys) async-staged to LDS (double buffered), shared by waves.
// Key loop unrolled x4 so the double-buffer phase is compile-time constant.
__global__ void __launch_bounds__(256) attn_kernel(
        const __bf16* __restrict__ Q,       // [64][1024][64] (pre-scaled)
        const __bf16* __restrict__ Kt,      // [64][1024][64]
        const __bf16* __restrict__ Vt,      // [64][64][1024]
        __bf16* __restrict__ Xo)            // [8192][512]
{
    __shared__ __align__(16) __bf16 Ks[2][32][64];    // 8 KB
    __shared__ __align__(16) __bf16 Vs[2][64][32];    // 8 KB
    __shared__ __align__(16) __bf16 lds_p[8][16][32]; // 8 KB per-wave P tiles

    const int lane = threadIdx.x & 31;
    const int wave = threadIdx.x >> 5;
    const int lh = (lane < 16) ? 0 : 1;
    const int bh = blockIdx.x >> 3;          // 0..63
    const int qb = (blockIdx.x & 7) * 128;   // 128-row query block
    const int qrow = qb + wave * 16 + (lane & 15);

    const __bf16* qbase = Q + ((bh << 10) + qrow) * DHEAD;
    const v16bf qf0 = load16(qbase +  0 + lh * 16, qbase +  8 + lh * 16);
    const v16bf qf1 = load16(qbase + 32 + lh * 16, qbase + 40 + lh * 16);

    const __bf16* Kb = Kt + ((size_t)bh << 16);
    const __bf16* Vb = Vt + ((size_t)bh << 16);

    // ---- issue first K/V block (kb = 0): 256 chunks each, 1 per thread ----
    {
        const int c = threadIdx.x;
        async_ld16((char*)(&Ks[0][0][0]) + c * 16,
                   Kb + (c >> 3) * DHEAD + (c & 7) * 8);  // K: [key][d]
        async_ld16((char*)(&Vs[0][0][0]) + c * 16,
                   Vb + ((c >> 2) << 10) + (c & 3) * 8);  // V: [d][key]
    }

    v8f o[4] = {};
    float m[8], l[8];
#pragma unroll
    for (int r = 0; r < 8; ++r) { m[r] = -1e30f; l[r] = 0.f; }

#pragma unroll 4
    for (int kb = 0; kb < NSEQ; kb += 32) {
        const int buf = (kb >> 5) & 1;
        wait_async0();
        __syncthreads();
        if (kb + 32 < NSEQ) {
            const int nb = buf ^ 1;
            const int k1 = kb + 32;
            const int c = threadIdx.x;
            async_ld16((char*)(&Ks[nb][0][0]) + c * 16,
                       Kb + (k1 + (c >> 3)) * DHEAD + (c & 7) * 8);
            async_ld16((char*)(&Vs[nb][0][0]) + c * 16,
                       Vb + ((c >> 2) << 10) + k1 + (c & 3) * 8);
        }
        // ---- S = Q @ K^T : load all K fragments first, then 4 WMMAs ----
        const __bf16* kp = &Ks[buf][lane & 15][lh * 16];
        const __bf16* kq = kp + 16 * DHEAD;
        const v16bf kf0 = load16(kp,      kp +  8);
        const v16bf kf1 = load16(kp + 32, kp + 40);
        const v16bf kf2 = load16(kq,      kq +  8);
        const v16bf kf3 = load16(kq + 32, kq + 40);
        v8f s0 = {}, s1 = {};
        s0 = wmma_bf16(qf0, kf0, s0);
        s0 = wmma_bf16(qf1, kf1, s0);
        s1 = wmma_bf16(qf0, kf2, s1);
        s1 = wmma_bf16(qf1, kf3, s1);
        // ---- online softmax (C layout: row = r + 8*lh, col = lane%16) ----
#pragma unroll
        for (int r = 0; r < 8; ++r) {
            const float bm = row_max16(fmaxf(s0[r], s1[r]));
            const float mn = fmaxf(m[r], bm);
            const float alpha = __expf(m[r] - mn);
            const float p0 = __expf(s0[r] - mn);
            const float p1 = __expf(s1[r] - mn);
            const float rs = row_sum16(p0 + p1);
            l[r] = l[r] * alpha + rs;
            m[r] = mn;
#pragma unroll
            for (int t = 0; t < 4; ++t) o[t][r] *= alpha;
            s0[r] = p0; s1[r] = p1;
        }
        // ---- C-layout -> A-layout through per-wave LDS tile ----
        {
            const int col = lane & 15;
            const int rb = lh * 8;
#pragma unroll
            for (int r = 0; r < 8; ++r) {
                lds_p[wave][rb + r][col]      = (__bf16)s0[r];
                lds_p[wave][rb + r][col + 16] = (__bf16)s1[r];
            }
        }
        __syncthreads();
        const __bf16* pp = &lds_p[wave][lane & 15][lh * 16];
        const v16bf pf = load16(pp, pp + 8);
        // ---- O += P @ V : load all V fragments first, then 4 WMMAs ----
        v16bf vf[4];
#pragma unroll
        for (int t = 0; t < 4; ++t) {
            const __bf16* vp = &Vs[buf][t * 16 + (lane & 15)][lh * 16];
            vf[t] = load16(vp, vp + 8);
        }
#pragma unroll
        for (int t = 0; t < 4; ++t) o[t] = wmma_bf16(pf, vf[t], o[t]);
    }

    // ---- normalize + store Xo[(B*1024+row)][h*64+d] ----
    const int B = bh >> 3, h = bh & 7;
#pragma unroll
    for (int t = 0; t < 4; ++t) {
        const int d = t * 16 + (lane & 15);
#pragma unroll
        for (int r = 0; r < 8; ++r) {
            const int row = qb + wave * 16 + r + lh * 8;
            Xo[((B << 10) + row) * DIM + h * DHEAD + d] = (__bf16)(o[t][r] / l[r]);
        }
    }
}

// ---------------------------------------------------------------------------
extern "C" void kernel_launch(void* const* d_in, const int* in_sizes, int n_in,
                              void* d_out, int out_size, void* d_ws, size_t ws_size,
                              hipStream_t stream) {
    const float* x     = (const float*)d_in[0];
    const float* Wqkv  = (const float*)d_in[1];
    const float* bqkv  = (const float*)d_in[2];
    const float* Wout  = (const float*)d_in[3];   // used as 512x512 (matches b_out)
    const float* bout  = (const float*)d_in[4];

    char* ws = (char*)d_ws;
    size_t off = 0;
    __bf16* xb  = (__bf16*)(ws + off); off += (size_t)MROWS * DIM * 2;
    __bf16* wqT = (__bf16*)(ws + off); off += (size_t)N_QKV * DIM * 2;
    __bf16* woT = (__bf16*)(ws + off); off += (size_t)DIM * DIM * 2;
    __bf16* Qb  = (__bf16*)(ws + off); off += (size_t)NBATCH * NHEAD * NSEQ * DHEAD * 2;
    __bf16* Kb  = (__bf16*)(ws + off); off += (size_t)NBATCH * NHEAD * NSEQ * DHEAD * 2;
    __bf16* Vtb = (__bf16*)(ws + off); off += (size_t)NBATCH * NHEAD * NSEQ * DHEAD * 2;
    __bf16* Xob = (__bf16*)(ws + off); off += (size_t)MROWS * DIM * 2;

    const int nx = MROWS * DIM;
    cvt_bf16_kernel<<<(nx + 255) / 256, 256, 0, stream>>>(x, xb, nx);

    const int nwq = DIM * N_QKV;
    transpose_bf16_kernel<<<(nwq + 255) / 256, 256, 0, stream>>>(Wqkv, wqT, DIM, N_QKV);

    const int nwo = DIM * DIM;
    transpose_bf16_kernel<<<(nwo + 255) / 256, 256, 0, stream>>>(Wout, woT, DIM, DIM);

    gemm_kernel<0><<<dim3(MROWS / 128, N_QKV / 128), 256, 0, stream>>>(
        xb, wqT, bqkv, Qb, Kb, Vtb, nullptr);

    attn_kernel<<<NBATCH * NHEAD * (NSEQ / 128), 256, 0, stream>>>(Qb, Kb, Vtb, Xob);

    gemm_kernel<1><<<dim3(MROWS / 128, DIM / 128), 256, 0, stream>>>(
        Xob, woT, bout, nullptr, nullptr, nullptr, (float*)d_out);
}
